// GILE_43250320670761
// MI455X (gfx1250) — matrix operand order repaired and checked by reading, not compiled, not run on hardware
//
#include <hip/hip_runtime.h>
#include <hip/hip_bf16.h>

// ---------------------------------------------------------------------------
// Types for CDNA5 WMMA (wave32): bf16 A/B operands are 16 x bf16 = 8 VGPRs,
// f32 C/D accumulator is 8 x f32 = 8 VGPRs.
// ---------------------------------------------------------------------------
typedef __attribute__((ext_vector_type(16))) __bf16 v16bf;
typedef __attribute__((ext_vector_type(8)))  __bf16 v8bf;
typedef __attribute__((ext_vector_type(8)))  float  v8f;

typedef unsigned short u16;

// Problem sizes (fixed by the reference)
#define DIM_N   4096      // rows of h / scores
#define DIM_M   32000     // rows of l / cols of scores
#define DIM_IN  1024      // input_dim
#define DIM_LB  512       // label_dim
#define DIM_J   512       // joint_dim

// GEMM tiling
#define BM   128
#define BN   128
#define BK   32           // one V_WMMA_F32_16X16X32_BF16 K-step
#define LDSK 40           // padded LDS row stride (elements) to dodge bank conflicts

__device__ __forceinline__ u16 f32_to_bf16_rne(float f) {
    unsigned u = __float_as_uint(f);
    u += 0x7fffu + ((u >> 16) & 1u);   // round-to-nearest-even
    return (u16)(u >> 16);
}

// Low 32 bits of a generic pointer to LDS == LDS byte offset (aperture in hi32).
__device__ __forceinline__ unsigned lds_off32(const void* p) {
    return (unsigned)(unsigned long long)p;
}

// CDNA5 async global->LDS copy: per-lane 16B, INST_OFFSET applies to both the
// LDS destination and the global source, so one base pair covers 32B/lane.
__device__ __forceinline__ void async_copy_32B(unsigned lds_byte, const u16* gptr) {
    asm volatile(
        "global_load_async_to_lds_b128 %0, %1, off\n\t"
        "global_load_async_to_lds_b128 %0, %1, off offset:16"
        :: "v"(lds_byte), "v"(gptr) : "memory");
}

__device__ __forceinline__ void wait_async0() {
    asm volatile("s_wait_asynccnt 0x0" ::: "memory");
}

// ---------------------------------------------------------------------------
// Elementwise f32 -> bf16 conversion (grid-stride)
// ---------------------------------------------------------------------------
__global__ __launch_bounds__(256) void cvt_bf16_kernel(const float* __restrict__ in,
                                                       u16* __restrict__ out, int n) {
    int i = blockIdx.x * blockDim.x + threadIdx.x;
    int stride = gridDim.x * blockDim.x;
    for (; i < n; i += stride) out[i] = f32_to_bf16_rne(in[i]);
}

// ---------------------------------------------------------------------------
// TN GEMM: C[M,N] = A[M,K](bf16, row-major) * B[N,K](bf16, row-major)^T
// EPI = 0: out_bf16 = bf16( relu(acc + bias[col]) * wvec[col] )   (gemm1)
// EPI = 1: out_bf16 = bf16( relu(acc + bias[col]) )               (gemm2)
// EPI = 2: out_f32  = acc + bscal[0]                              (gemm3)
// 256 threads = 8 waves (wave32), waves tiled 4(M) x 2(N), each wave owns a
// 32x64 tile = 2x4 frags. Double-buffered LDS fed by async global->LDS DMA.
// ---------------------------------------------------------------------------
template<int EPI>
__global__ __launch_bounds__(256) void gemm_bf16_tn_kernel(
    const u16* __restrict__ A, const u16* __restrict__ Bm,
    int M, int N, int K,
    const float* __restrict__ bias,  const float* __restrict__ wvec,
    const float* __restrict__ bscal,
    u16* __restrict__ outB, float* __restrict__ outF)
{
    __shared__ u16 Al[2][BM * LDSK];
    __shared__ u16 Bl[2][BN * LDSK];

    const int tid   = threadIdx.x;
    const int lane  = tid & 31;
    const int wave  = tid >> 5;        // 0..7
    const int waveM = wave >> 1;       // 0..3 -> which 32-row slab
    const int waveN = wave & 1;        // 0..1 -> which 64-col slab
    const int half  = lane >> 4;       // 0/1: lane group
    const int l16   = lane & 15;

    const int blockN0 = blockIdx.x * BN;
    const int blockM0 = blockIdx.y * BM;

    v8f acc[2][4];
#pragma unroll
    for (int mf = 0; mf < 2; ++mf)
#pragma unroll
        for (int nf = 0; nf < 4; ++nf)
            acc[mf][nf] = (v8f)(0.0f);

    // tile loader: each thread DMAs 32 bytes of A-tile and 32 bytes of B-tile
    const int lrow = tid >> 1;             // 0..127
    const int lcol = (tid & 1) * 16;       // element offset 0 or 16 within BK

    const u16* gA = A  + (size_t)(blockM0 + lrow) * K + lcol;
    const u16* gB = Bm + (size_t)(blockN0 + lrow) * K + lcol;
    const unsigned ldsA[2] = { lds_off32(&Al[0][lrow * LDSK + lcol]),
                               lds_off32(&Al[1][lrow * LDSK + lcol]) };
    const unsigned ldsB[2] = { lds_off32(&Bl[0][lrow * LDSK + lcol]),
                               lds_off32(&Bl[1][lrow * LDSK + lcol]) };

    // prologue: DMA tile 0 into buffer 0
    async_copy_32B(ldsA[0], gA);
    async_copy_32B(ldsB[0], gB);
    wait_async0();
    __syncthreads();

    const int steps = K / BK;
    for (int s = 0; s < steps; ++s) {
        const int buf = s & 1;

        // kick off DMA for the next tile into the other buffer (safe: barrier
        // at the end of iteration s-1 retired all reads of that buffer)
        if (s + 1 < steps) {
            const int k1 = (s + 1) * BK;
            async_copy_32B(ldsA[buf ^ 1], gA + k1);
            async_copy_32B(ldsB[buf ^ 1], gB + k1);
        }

        // ---- assemble WMMA fragments per CDNA5 VGPR layouts ----
        // A (16x32): lanes 0-15 hold row m=l16 with K {0..7,16..23};
        //            lanes 16-31 hold K {8..15,24..31}
        v16bf afrag[2];
#pragma unroll
        for (int mf = 0; mf < 2; ++mf) {
            const int mrow = waveM * 32 + mf * 16 + l16;
            v8bf lo = *(const v8bf*)&Al[buf][mrow * LDSK + 8 * half];
            v8bf hi = *(const v8bf*)&Al[buf][mrow * LDSK + 16 + 8 * half];
            afrag[mf] = __builtin_shufflevector(lo, hi,
                0, 1, 2, 3, 4, 5, 6, 7, 8, 9, 10, 11, 12, 13, 14, 15);
        }
        // B (32x16): same K order as A: lanes 0-15 K {0..7,16..23}, lanes 16-31 K {8..15,24..31}
        v16bf bfrag[4];
#pragma unroll
        for (int nf = 0; nf < 4; ++nf) {
            const int ncol = waveN * 64 + nf * 16 + l16;
            v8bf lo = *(const v8bf*)&Bl[buf][ncol * LDSK + 8 * half];
            v8bf hi = *(const v8bf*)&Bl[buf][ncol * LDSK + 16 + 8 * half];
            bfrag[nf] = __builtin_shufflevector(lo, hi,
                0, 1, 2, 3, 4, 5, 6, 7, 8, 9, 10, 11, 12, 13, 14, 15);
        }

        // ---- 8x V_WMMA_F32_16X16X32_BF16 (overlaps the in-flight DMA) ----
#pragma unroll
        for (int mf = 0; mf < 2; ++mf)
#pragma unroll
            for (int nf = 0; nf < 4; ++nf)
                acc[mf][nf] = __builtin_amdgcn_wmma_f32_16x16x32_bf16(
                    /*neg_a=*/false, afrag[mf], /*neg_b=*/false, bfrag[nf],
                    /*c_mod=*/(short)0, acc[mf][nf],
                    /*reuse_a=*/false, /*reuse_b=*/false);

        // retire next-tile DMA, then block-wide barrier before swapping buffers
        wait_async0();
        __syncthreads();
    }

    // ---- epilogue ----
    // C/D layout: VGPR r holds row (r + 8*half), col = l16 (within the frag)
    const float bconst = (EPI == 2) ? bscal[0] : 0.0f;
#pragma unroll
    for (int mf = 0; mf < 2; ++mf) {
#pragma unroll
        for (int nf = 0; nf < 4; ++nf) {
            const int col = blockN0 + waveN * 64 + nf * 16 + l16;
            float bv = 0.0f, wv = 0.0f;
            if (EPI == 0) { bv = bias[col]; wv = wvec[col]; }
            if (EPI == 1) { bv = bias[col]; }
#pragma unroll
            for (int r = 0; r < 8; ++r) {
                const int row = blockM0 + waveM * 32 + mf * 16 + 8 * half + r;
                float v = acc[mf][nf][r];
                if (EPI == 0) {
                    v += bv; v = v > 0.0f ? v * wv : 0.0f;
                    outB[(size_t)row * N + col] = f32_to_bf16_rne(v);
                } else if (EPI == 1) {
                    v += bv; v = v > 0.0f ? v : 0.0f;
                    outB[(size_t)row * N + col] = f32_to_bf16_rne(v);
                } else {
                    outF[(size_t)row * N + col] = v + bconst;
                }
            }
        }
    }
}

// ---------------------------------------------------------------------------
// Host-side launcher
// ---------------------------------------------------------------------------
static inline size_t align256(size_t x) { return (x + 255) & ~(size_t)255; }

extern "C" void kernel_launch(void* const* d_in, const int* in_sizes, int n_in,
                              void* d_out, int out_size, void* d_ws, size_t ws_size,
                              hipStream_t stream) {
    (void)in_sizes; (void)n_in; (void)out_size; (void)ws_size;

    const float* h  = (const float*)d_in[0];  // [4096,1024]
    const float* l  = (const float*)d_in[1];  // [32000,512]
    const float* Wh = (const float*)d_in[2];  // [512,1024]
    const float* bh = (const float*)d_in[3];  // [512]
    const float* Wl = (const float*)d_in[4];  // [512,512]
    const float* bl = (const float*)d_in[5];  // [512]
    const float* w  = (const float*)d_in[6];  // [512]
    const float* b  = (const float*)d_in[7];  // [1]
    float* out = (float*)d_out;               // [4096,32000]

    // workspace carve-up (bf16 staging buffers)
    char* p = (char*)d_ws;
    u16* hB  = (u16*)p; p += align256((size_t)DIM_N  * DIM_IN * 2);
    u16* WhB = (u16*)p; p += align256((size_t)DIM_J  * DIM_IN * 2);
    u16* lB  = (u16*)p; p += align256((size_t)DIM_M  * DIM_LB * 2);
    u16* WlB = (u16*)p; p += align256((size_t)DIM_J  * DIM_LB * 2);
    u16* haB = (u16*)p; p += align256((size_t)DIM_N  * DIM_J  * 2);  // (relu(h Wh^T + bh) * w) bf16
    u16* laB = (u16*)p; p += align256((size_t)DIM_M  * DIM_J  * 2);  // relu(l Wl^T + bl) bf16

    // 1) convert inputs to bf16
    cvt_bf16_kernel<<<2048, 256, 0, stream>>>(h,  hB,  DIM_N * DIM_IN);
    cvt_bf16_kernel<<<512,  256, 0, stream>>>(Wh, WhB, DIM_J * DIM_IN);
    cvt_bf16_kernel<<<4096, 256, 0, stream>>>(l,  lB,  DIM_M * DIM_LB);
    cvt_bf16_kernel<<<256,  256, 0, stream>>>(Wl, WlB, DIM_J * DIM_LB);

    // 2) haB = bf16( relu(h @ Wh^T + bh) * w )      [4096 x 512], K=1024
    {
        dim3 grid(DIM_J / BN, DIM_N / BM);
        gemm_bf16_tn_kernel<0><<<grid, 256, 0, stream>>>(
            hB, WhB, DIM_N, DIM_J, DIM_IN, bh, w, nullptr, haB, nullptr);
    }
    // 3) laB = bf16( relu(l @ Wl^T + bl) )          [32000 x 512], K=512
    {
        dim3 grid(DIM_J / BN, DIM_M / BM);
        gemm_bf16_tn_kernel<1><<<grid, 256, 0, stream>>>(
            lB, WlB, DIM_M, DIM_J, DIM_LB, bl, nullptr, nullptr, laB, nullptr);
    }
    // 4) out = haB @ laB^T + b                      [4096 x 32000], K=512
    {
        dim3 grid(DIM_M / BN, DIM_N / BM);
        gemm_bf16_tn_kernel<2><<<grid, 256, 0, stream>>>(
            haB, laB, DIM_N, DIM_M, DIM_J, nullptr, nullptr, b, nullptr, out);
    }
}
